// MultiHeadAttentionLayer_36704790512332
// MI455X (gfx1250) — matrix operand order repaired and checked by reading, not compiled, not run on hardware
//
#include <hip/hip_runtime.h>
#include <hip/hip_bf16.h>

// Problem constants (match reference)
#define BSZ   4
#define LSEQ  2048
#define DMOD  1024
#define NHEAD 16
#define HDIM  64
#define MROWS (BSZ * LSEQ)        // 8192

typedef __attribute__((ext_vector_type(16))) __bf16 v16bf;
typedef __attribute__((ext_vector_type(8)))  __bf16 bf16x8;
typedef __attribute__((ext_vector_type(8)))  float  v8f;

union U16v { v16bf v; bf16x8 h[2]; };

__device__ inline __bf16 to_bf16(float f) {
  unsigned u = __builtin_bit_cast(unsigned, f);
  u += 0x7FFFu + ((u >> 16) & 1u);            // round-to-nearest-even
  unsigned short hs = (unsigned short)(u >> 16);
  return __builtin_bit_cast(__bf16, hs);
}

__device__ inline v8f wmma_bf16(v16bf a, v16bf b, v8f c) {
  return __builtin_amdgcn_wmma_f32_16x16x32_bf16(false, a, false, b, (short)0, c,
                                                 false, false);
}

// ---------------- fp32 -> bf16 elementwise ----------------
__global__ void cvt_f32_bf16(const float* __restrict__ in, __bf16* __restrict__ out, int n) {
  int i = blockIdx.x * blockDim.x + threadIdx.x;
  if (i < n) out[i] = to_bf16(in[i]);
}

// ---------------- W [K=1024, N=1024] fp32 -> Wt [N, K] bf16 ----------------
__global__ void transpose_cvt(const float* __restrict__ in, __bf16* __restrict__ out) {
  int idx = blockIdx.x * 256 + threadIdx.x;   // over N*K
  int n = idx >> 10;
  int d = idx & 1023;
  out[idx] = to_bf16(in[d * 1024 + n]);
}

// ---------------- WMMA GEMM: one wave = 32x64 output tile ----------------
// C[m,n] = A[m,:] . Bt[n,:] + bias[n]
// MODE 0: Q -> bf16 [B,H,L,64], scaled by 1/sqrt(64)
// MODE 1: K -> bf16 [B,H,L,64]
// MODE 2: V -> bf16 [B,H,64,L]   (transposed for P.V B-fragments)
// MODE 3: fp32 out = acc + bias + residual   (row-major [M,N])
template <int MODE>
__global__ void __launch_bounds__(256) gemm_wmma(
    const __bf16* __restrict__ A, const __bf16* __restrict__ Bt,
    const float* __restrict__ bias, void* __restrict__ Out,
    const float* __restrict__ Resid, float scale) {
  const int lane = threadIdx.x & 31;
  const int wave = threadIdx.x >> 5;
  const int hh   = lane >> 4;      // lane half
  const int ln   = lane & 15;
  const int wid  = blockIdx.x * 8 + wave;   // over (M/32)*(N/64) = 4096 tiles
  const int tm   = wid >> 4;       // 256 M-tiles of 32
  const int tn   = wid & 15;       // 16  N-tiles of 64
  const int m0   = tm << 5;
  const int n0   = tn << 6;

  const __bf16* arow0 = A + (size_t)(m0 + ln) * DMOD;
  const __bf16* arow1 = A + (size_t)(m0 + 16 + ln) * DMOD;
  const __bf16* brow  = Bt + (size_t)(n0 + ln) * DMOD;

  v8f acc[8] = {};   // acc[i*4+j]: i = M-subtile (0..1), j = N-subtile (0..3)
  for (int kk = 0; kk < DMOD; kk += 32) {
    U16v a0, a1;
    a0.h[0] = *(const bf16x8*)(arow0 + kk + 16 * hh);
    a0.h[1] = *(const bf16x8*)(arow0 + kk + 16 * hh + 8);
    a1.h[0] = *(const bf16x8*)(arow1 + kk + 16 * hh);
    a1.h[1] = *(const bf16x8*)(arow1 + kk + 16 * hh + 8);
    __builtin_prefetch(arow0 + kk + 32, 0, 1);
    __builtin_prefetch(arow1 + kk + 32, 0, 1);
#pragma unroll
    for (int j = 0; j < 4; ++j) {
      v16bf b = *(const v16bf*)(brow + (size_t)j * 16 * DMOD + kk + 16 * hh);
      __builtin_prefetch(brow + (size_t)j * 16 * DMOD + kk + 32, 0, 1);
      acc[j]     = wmma_bf16(a0.v, b, acc[j]);
      acc[4 + j] = wmma_bf16(a1.v, b, acc[4 + j]);
    }
  }

#pragma unroll
  for (int i = 0; i < 2; ++i) {
#pragma unroll
    for (int j = 0; j < 4; ++j) {
#pragma unroll
      for (int e = 0; e < 8; ++e) {
        const int m = m0 + i * 16 + e + 8 * hh;   // global row (b*L + l)
        const int n = n0 + j * 16 + ln;           // global col
        float val = acc[i * 4 + j][e] + bias[n];
        if (MODE == 0 || MODE == 1) {
          const int b_ = m / LSEQ, l_ = m % LSEQ, h_ = n >> 6, d_ = n & 63;
          ((__bf16*)Out)[(((size_t)(b_ * NHEAD + h_) * LSEQ + l_) << 6) + d_] =
              to_bf16(val * scale);
        } else if (MODE == 2) {
          const int b_ = m / LSEQ, l_ = m % LSEQ, h_ = n >> 6, d_ = n & 63;
          ((__bf16*)Out)[((size_t)(b_ * NHEAD + h_) * HDIM + d_) * LSEQ + l_] =
              to_bf16(val);
        } else {
          const size_t idx = (size_t)m * DMOD + n;
          ((float*)Out)[idx] = val + Resid[idx];
        }
      }
    }
  }
}

// ---------------- Flash attention: one wave = 16 query rows, 64-key chunks ----------------
__global__ void __launch_bounds__(256) attn_kernel(
    const __bf16* __restrict__ q,   // [B,H,L,64], pre-scaled by 1/sqrt(64)
    const __bf16* __restrict__ k,   // [B,H,L,64]
    const __bf16* __restrict__ vt,  // [B,H,64,L]
    __bf16* __restrict__ attnO) {   // [B*L, 1024] bf16
  __shared__ __attribute__((aligned(16))) __bf16 plds[8][16][64];

  const int lane = threadIdx.x & 31;
  const int wave = threadIdx.x >> 5;
  const int hh   = lane >> 4;
  const int ln   = lane & 15;

  int bid = blockIdx.x;                 // b * 16(heads) * 16(qtiles of 128)
  const int qt = bid & 15; bid >>= 4;
  const int hd = bid & 15; bid >>= 4;
  const int bb = bid;
  const int q0 = qt * 128 + wave * 16;

  const __bf16* qh = q  + (size_t)(bb * NHEAD + hd) * LSEQ * HDIM;
  const __bf16* kh = k  + (size_t)(bb * NHEAD + hd) * LSEQ * HDIM;
  const __bf16* vh = vt + (size_t)(bb * NHEAD + hd) * HDIM * LSEQ;

  // Q tile A-fragments for head-dim chunks 0..31 and 32..63
  U16v aq0, aq1;
  {
    const __bf16* qr = qh + (size_t)(q0 + ln) * HDIM;
    aq0.h[0] = *(const bf16x8*)(qr + 16 * hh);
    aq0.h[1] = *(const bf16x8*)(qr + 16 * hh + 8);
    aq1.h[0] = *(const bf16x8*)(qr + 32 + 16 * hh);
    aq1.h[1] = *(const bf16x8*)(qr + 32 + 16 * hh + 8);
  }

  v8f o[4] = {};                    // 16x64 output accumulator (4 vd tiles)
  float mrow[8], lrow[8];
#pragma unroll
  for (int e = 0; e < 8; ++e) { mrow[e] = -3.0e38f; lrow[e] = 0.0f; }

  for (int kb = 0; kb < LSEQ; kb += 64) {
    // ---- S = Q . K^T : four 16x16 key tiles (64 keys), head-dim in 2 chunks
    v8f s[4];
#pragma unroll
    for (int t = 0; t < 4; ++t) {
      const __bf16* kr = kh + (size_t)(kb + t * 16 + ln) * HDIM + 16 * hh;
      v16bf bk0 = *(const v16bf*)(kr);
      v16bf bk1 = *(const v16bf*)(kr + 32);
      v8f z = {};
      z = wmma_bf16(aq0.v, bk0, z);
      s[t] = wmma_bf16(aq1.v, bk1, z);
    }

    // ---- Online softmax; D-fragment row m = e + 8*hh, col = t*16 + ln
#pragma unroll
    for (int e = 0; e < 8; ++e) {
      float cm = fmaxf(fmaxf(s[0][e], s[1][e]), fmaxf(s[2][e], s[3][e]));
#pragma unroll
      for (int off = 1; off < 16; off <<= 1) cm = fmaxf(cm, __shfl_xor(cm, off, 32));
      const float nm    = fmaxf(mrow[e], cm);
      const float alpha = __expf(mrow[e] - nm);
      const float p0 = __expf(s[0][e] - nm);
      const float p1 = __expf(s[1][e] - nm);
      const float p2 = __expf(s[2][e] - nm);
      const float p3 = __expf(s[3][e] - nm);
      float ls = (p0 + p1) + (p2 + p3);
#pragma unroll
      for (int off = 1; off < 16; off <<= 1) ls += __shfl_xor(ls, off, 32);
      lrow[e] = lrow[e] * alpha + ls;
      mrow[e] = nm;
      o[0][e] *= alpha; o[1][e] *= alpha; o[2][e] *= alpha; o[3][e] *= alpha;
      const int m = e + 8 * hh;
      plds[wave][m][ln]      = to_bf16(p0);
      plds[wave][m][16 + ln] = to_bf16(p1);
      plds[wave][m][32 + ln] = to_bf16(p2);
      plds[wave][m][48 + ln] = to_bf16(p3);
    }
    __syncthreads();   // order LDS writes before A-layout reads

    // ---- Re-read P as two 16x32 A-fragments (key cols 0..31 and 32..63)
    U16v pa0, pa1;
    pa0.h[0] = *(const bf16x8*)&plds[wave][ln][16 * hh];
    pa0.h[1] = *(const bf16x8*)&plds[wave][ln][16 * hh + 8];
    pa1.h[0] = *(const bf16x8*)&plds[wave][ln][32 + 16 * hh];
    pa1.h[1] = *(const bf16x8*)&plds[wave][ln][32 + 16 * hh + 8];

    // ---- O += P . V ; V B-fragments from transposed V, keys contiguous
#pragma unroll
    for (int j = 0; j < 4; ++j) {
      const __bf16* vr = vh + (size_t)(j * 16 + ln) * LSEQ + kb + 16 * hh;
      v16bf bv0 = *(const v16bf*)(vr);
      v16bf bv1 = *(const v16bf*)(vr + 32);
      o[j] = wmma_bf16(pa0.v, bv0, o[j]);
      o[j] = wmma_bf16(pa1.v, bv1, o[j]);
    }
    __syncthreads();
  }

  // ---- Normalize and store bf16 attention output [B*L, H*64]
#pragma unroll
  for (int e = 0; e < 8; ++e) {
    const float inv = 1.0f / lrow[e];
    const int   m   = q0 + e + 8 * hh;
    const size_t base = ((size_t)bb * LSEQ + m) * DMOD + hd * HDIM + ln;
    attnO[base]      = to_bf16(o[0][e] * inv);
    attnO[base + 16] = to_bf16(o[1][e] * inv);
    attnO[base + 32] = to_bf16(o[2][e] * inv);
    attnO[base + 48] = to_bf16(o[3][e] * inv);
  }
}

// ---------------- LayerNorm over rows of 1024 ----------------
__global__ void __launch_bounds__(256) layernorm_kernel(
    const float* __restrict__ res, const float* __restrict__ gamma,
    const float* __restrict__ beta, float* __restrict__ out) {
  __shared__ float red[2][8];
  const int row = blockIdx.x;
  const float* x = res + (size_t)row * DMOD;
  float s = 0.f, ss = 0.f;
  for (int i = threadIdx.x; i < DMOD; i += 256) {
    const float v = x[i];
    s += v; ss += v * v;
  }
#pragma unroll
  for (int off = 1; off < 32; off <<= 1) {
    s += __shfl_xor(s, off, 32);
    ss += __shfl_xor(ss, off, 32);
  }
  const int wave = threadIdx.x >> 5, lane = threadIdx.x & 31;
  if (lane == 0) { red[0][wave] = s; red[1][wave] = ss; }
  __syncthreads();
  if (wave == 0) {
    s  = (lane < 8) ? red[0][lane] : 0.f;
    ss = (lane < 8) ? red[1][lane] : 0.f;
#pragma unroll
    for (int off = 1; off < 8; off <<= 1) {
      s += __shfl_xor(s, off, 32);
      ss += __shfl_xor(ss, off, 32);
    }
    if (lane == 0) { red[0][0] = s; red[1][0] = ss; }
  }
  __syncthreads();
  const float mean = red[0][0] * (1.0f / DMOD);
  const float var  = red[1][0] * (1.0f / DMOD) - mean * mean;
  const float rstd = rsqrtf(var + 1e-5f);
  for (int i = threadIdx.x; i < DMOD; i += 256) {
    out[(size_t)row * DMOD + i] = (x[i] - mean) * rstd * gamma[i] + beta[i];
  }
}

extern "C" void kernel_launch(void* const* d_in, const int* in_sizes, int n_in,
                              void* d_out, int out_size, void* d_ws, size_t ws_size,
                              hipStream_t stream) {
  const float* queries = (const float*)d_in[0];
  const float* Wq = (const float*)d_in[1];
  const float* bq = (const float*)d_in[2];
  const float* Wk = (const float*)d_in[3];
  const float* bk = (const float*)d_in[4];
  const float* Wv = (const float*)d_in[5];
  const float* bv = (const float*)d_in[6];
  const float* Wo = (const float*)d_in[7];
  const float* bo = (const float*)d_in[8];
  const float* gamma = (const float*)d_in[9];
  const float* beta  = (const float*)d_in[10];

  // Workspace layout (bytes). attnO aliases Xb (dead after V gemm);
  // res aliases q+k (dead after attention). Peak ~72 MB.
  char* ws = (char*)d_ws;
  const size_t SZ_X  = (size_t)MROWS * DMOD * 2;   // 16 MB bf16
  const size_t SZ_W  = (size_t)DMOD * DMOD * 2;    //  2 MB bf16
  const size_t SZ_QK = (size_t)BSZ * NHEAD * LSEQ * HDIM * 2; // 16 MB bf16

  __bf16* Xb    = (__bf16*)(ws);
  __bf16* attnO = (__bf16*)(ws);                   // alias of Xb
  __bf16* Wqt   = (__bf16*)(ws + SZ_X);
  __bf16* Wkt   = (__bf16*)(ws + SZ_X + SZ_W);
  __bf16* Wvt   = (__bf16*)(ws + SZ_X + 2 * SZ_W);
  __bf16* Wot   = (__bf16*)(ws + SZ_X + 3 * SZ_W);
  __bf16* qbuf  = (__bf16*)(ws + SZ_X + 4 * SZ_W);
  __bf16* kbuf  = (__bf16*)(ws + SZ_X + 4 * SZ_W + SZ_QK);
  float*  res   = (float*)(ws + SZ_X + 4 * SZ_W);  // alias of q+k (32 MB f32)
  __bf16* vtbuf = (__bf16*)(ws + SZ_X + 4 * SZ_W + 2 * SZ_QK);

  const int nX = MROWS * DMOD;                     // 8,388,608
  cvt_f32_bf16<<<nX / 256, 256, 0, stream>>>(queries, Xb, nX);
  transpose_cvt<<<(DMOD * DMOD) / 256, 256, 0, stream>>>(Wq, Wqt);
  transpose_cvt<<<(DMOD * DMOD) / 256, 256, 0, stream>>>(Wk, Wkt);
  transpose_cvt<<<(DMOD * DMOD) / 256, 256, 0, stream>>>(Wv, Wvt);
  transpose_cvt<<<(DMOD * DMOD) / 256, 256, 0, stream>>>(Wo, Wot);

  const int gemm_blocks = (MROWS / 32) * (DMOD / 64) / 8;   // 512
  gemm_wmma<0><<<gemm_blocks, 256, 0, stream>>>(Xb, Wqt, bq, qbuf,  nullptr, 0.125f);
  gemm_wmma<1><<<gemm_blocks, 256, 0, stream>>>(Xb, Wkt, bk, kbuf,  nullptr, 1.0f);
  gemm_wmma<2><<<gemm_blocks, 256, 0, stream>>>(Xb, Wvt, bv, vtbuf, nullptr, 1.0f);

  attn_kernel<<<BSZ * NHEAD * (LSEQ / 128), 256, 0, stream>>>(qbuf, kbuf, vtbuf, attnO);

  gemm_wmma<3><<<gemm_blocks, 256, 0, stream>>>(attnO, Wot, bo, res, queries, 1.0f);

  layernorm_kernel<<<MROWS, 256, 0, stream>>>(res, gamma, beta, (float*)d_out);
}
